// Model_23330262351982
// MI455X (gfx1250) — matrix operand order repaired and checked by reading, not compiled, not run on hardware
//
#include <hip/hip_runtime.h>
#include <hip/hip_bf16.h>

#define NUU 50000
#define NII 30000
#define DDIM 64
#define NTOT (NUU + NII)
#define BBATCH 1024

typedef __attribute__((ext_vector_type(16))) _Float16 v16h;
typedef __attribute__((ext_vector_type(8)))  float    v8f;

// ---------------------------------------------------------------------------
// init (float4-vectorized): x = concat(emb_user,emb_item)(+beh), acc=x, y=0
// ---------------------------------------------------------------------------
__global__ void init_kernel(const float* __restrict__ eu, const float* __restrict__ ei,
                            const float* __restrict__ beh,
                            float* __restrict__ x, float* __restrict__ acc,
                            float* __restrict__ y) {
    int i = blockIdx.x * blockDim.x + threadIdx.x;       // over NTOT*16 float4s
    if (i >= NTOT * (DDIM / 4)) return;
    int node = i >> 4;
    int d4 = (i & 15) * 4;
    size_t fo = (size_t)i * 4;
    float4 v = (node < NUU) ? *(const float4*)(eu + fo)
                            : *(const float4*)(ei + fo - (size_t)NUU * DDIM);
    if (beh) {
        float4 bv = *(const float4*)(beh + d4);
        v.x += bv.x; v.y += bv.y; v.z += bv.z; v.w += bv.w;
    }
    *(float4*)(x + fo) = v;
    *(float4*)(acc + fo) = v;
    *(float4*)(y + fo) = make_float4(0.f, 0.f, 0.f, 0.f);
}

// ---------------------------------------------------------------------------
// COO SpMM scatter: y[r,:] += v * x[c,:].  16 threads/edge, float4 each.
// x (20.5MB) fits in L2; atomics resolve in L2.
// ---------------------------------------------------------------------------
__global__ void spmm_kernel(const int* __restrict__ rows, const int* __restrict__ cols,
                            const float* __restrict__ vals, int E,
                            const float* __restrict__ x, float* __restrict__ y) {
    long long gid = (long long)blockIdx.x * blockDim.x + threadIdx.x;
    int e = (int)(gid >> 4);
    if (e >= E) return;
    int q = (int)(gid & 15);
    int r = rows[e];
    int c = cols[e];
    float v = vals[e];
    const float4 xv = *(const float4*)(x + (size_t)c * DDIM + q * 4);
    float* yp = y + (size_t)r * DDIM + q * 4;
    unsafeAtomicAdd(yp + 0, v * xv.x);
    unsafeAtomicAdd(yp + 1, v * xv.y);
    unsafeAtomicAdd(yp + 2, v * xv.z);
    unsafeAtomicAdd(yp + 3, v * xv.w);
}

// acc += y ; optionally zero old-x buffer (becomes next layer's y). float4.
__global__ void acc_add_zero_kernel(float* __restrict__ acc, const float* __restrict__ y,
                                    float* __restrict__ xold, int doZero) {
    int i = blockIdx.x * blockDim.x + threadIdx.x;
    if (i >= NTOT * (DDIM / 4)) return;
    size_t fo = (size_t)i * 4;
    float4 a = *(const float4*)(acc + fo);
    float4 b = *(const float4*)(y + fo);
    a.x += b.x; a.y += b.y; a.z += b.z; a.w += b.w;
    *(float4*)(acc + fo) = a;
    if (doZero) *(float4*)(xold + fo) = make_float4(0.f, 0.f, 0.f, 0.f);
}

// ubraw[b,:] = acc[batch_users[b],:]/4 ; ibraw[i,:] = acc[NU+i,:]/4  (float4)
__global__ void extract_kernel(const float* __restrict__ acc, const int* __restrict__ bu,
                               float* __restrict__ ubraw, float* __restrict__ ibraw) {
    int i = blockIdx.x * blockDim.x + threadIdx.x;   // over (B+NI)*16 float4s
    const int nu4 = BBATCH * (DDIM / 4);
    const int nt4 = (BBATCH + NII) * (DDIM / 4);
    if (i >= nt4) return;
    if (i < nu4) {
        int b = i >> 4, d4 = (i & 15) * 4;
        float4 v = *(const float4*)(acc + (size_t)bu[b] * DDIM + d4);
        v.x *= 0.25f; v.y *= 0.25f; v.z *= 0.25f; v.w *= 0.25f;
        *(float4*)(ubraw + (size_t)i * 4) = v;
    } else {
        size_t j = (size_t)(i - nu4) * 4;
        float4 v = *(const float4*)(acc + (size_t)NUU * DDIM + j);
        v.x *= 0.25f; v.y *= 0.25f; v.z *= 0.25f; v.w *= 0.25f;
        *(float4*)(ibraw + j) = v;
    }
}

// ---------------------------------------------------------------------------
// user gate: g=sigmoid(a@W.T+b); ub=g*a+(1-g)*bb; L2-normalize; fold in w;
// emit f16 GEMM operand. One 64-thread block per batch row.
// ---------------------------------------------------------------------------
__global__ void gate_user_kernel(const float* __restrict__ a_all, const float* __restrict__ b_beh,
                                 const float* __restrict__ W, const float* __restrict__ bias,
                                 const float* __restrict__ wvec, _Float16* __restrict__ out) {
    __shared__ float sa[DDIM], sb[DDIM], sval[DDIM], sred[DDIM];
    int r = blockIdx.x;
    int d = threadIdx.x;
    sa[d] = a_all[r * DDIM + d];
    sb[d] = b_beh[r * DDIM + d];
    __syncthreads();
    float dot = 0.0f;
#pragma unroll
    for (int k = 0; k < DDIM; ++k) dot += sa[k] * W[d * DDIM + k];
    float g = 1.0f / (1.0f + __expf(-(dot + bias[d])));
    float v = g * sa[d] + (1.0f - g) * sb[d];
    sval[d] = v;
    sred[d] = v * v;
    __syncthreads();
    for (int s = 32; s > 0; s >>= 1) {
        if (d < s) sred[d] += sred[d + s];
        __syncthreads();
    }
    float inv = 1.0f / (sqrtf(sred[0]) + 1e-12f);
    out[r * DDIM + d] = (_Float16)(sval[d] * inv * wvec[d]);
}

// item gate: same gating, no norm/weight. One 64-thread block per item row.
__global__ void gate_item_kernel(const float* __restrict__ a_all, const float* __restrict__ b_beh,
                                 const float* __restrict__ W, const float* __restrict__ bias,
                                 _Float16* __restrict__ out) {
    __shared__ float sa[DDIM], sb[DDIM];
    int r = blockIdx.x;
    int d = threadIdx.x;
    sa[d] = a_all[r * DDIM + d];
    sb[d] = b_beh[r * DDIM + d];
    __syncthreads();
    float dot = 0.0f;
#pragma unroll
    for (int k = 0; k < DDIM; ++k) dot += sa[k] * W[d * DDIM + k];
    float g = 1.0f / (1.0f + __expf(-(dot + bias[d])));
    out[r * DDIM + d] = (_Float16)(g * sa[d] + (1.0f - g) * sb[d]);
}

// ---------------------------------------------------------------------------
// C[1024,30000] = relu(A[1024,64] @ Bm[30000,64]^T), C[:,NII-1]=0.
// One wave per 16x16 tile; K=64 -> 2x v_wmma_f32_16x16x32_f16.
// NII compile-time: row stride 120000B fits imm24 -> single base + 8 offset
// stores. Nontemporal stores: output is write-once, keep L2 for operands.
// ---------------------------------------------------------------------------
__global__ __launch_bounds__(256) void wmma_gemm_relu_kernel(
    const _Float16* __restrict__ A, const _Float16* __restrict__ Bm,
    float* __restrict__ C) {
    int wave = threadIdx.x >> 5;
    int lane = threadIdx.x & 31;
    int ct = blockIdx.x * 8 + wave;          // column tile (16 cols)
    int rt = blockIdx.y;                     // row tile (16 rows)
    const int nTilesC = NII / 16;            // 1875
    if (ct >= nTilesC) return;               // wave-uniform: EXEC all-ones for WMMA

    int half = lane >> 4;
    int l15 = lane & 15;

    // ---- A fragments (row rt*16+l15), CDNA5 16-bit A 16x32 layout ----
    const _Float16* arow = A + (size_t)(rt * 16 + l15) * DDIM;
    int kb = half * 8;
    v16h a0, a1;
    ((float4*)&a0)[0] = *(const float4*)(arow + kb);
    ((float4*)&a0)[1] = *(const float4*)(arow + 16 + kb);
    ((float4*)&a1)[0] = *(const float4*)(arow + 32 + kb);
    ((float4*)&a1)[1] = *(const float4*)(arow + 48 + kb);

    // ---- B fragments (col ct*16+l15), CDNA5 16-bit B 32x16 layout ----
    const _Float16* brow = Bm + (size_t)(ct * 16 + l15) * DDIM;
    int koff = half * 8;
    v16h b0, b1;
    ((float4*)&b0)[0] = *(const float4*)(brow + koff);
    ((float4*)&b0)[1] = *(const float4*)(brow + 16 + koff);
    ((float4*)&b1)[0] = *(const float4*)(brow + 32 + koff);
    ((float4*)&b1)[1] = *(const float4*)(brow + 48 + koff);

    v8f c = {};
    c = __builtin_amdgcn_wmma_f32_16x16x32_f16(false, a0, false, b0, (short)0, c, false, false);
    c = __builtin_amdgcn_wmma_f32_16x16x32_f16(false, a1, false, b1, (short)0, c, false, false);

    int col = ct * 16 + l15;
    float maskScale = (col == NII - 1) ? 0.0f : 1.0f;   // zero last item column
    // single base pointer; per-row stride folds into imm24 store offsets
    float* cbase = C + (size_t)(rt * 16 + half * 8) * NII + col;
#pragma unroll
    for (int v = 0; v < 8; ++v) {
        float x = fmaxf(c[v], 0.0f) * maskScale;
        __builtin_nontemporal_store(x, cbase + (size_t)v * NII);
    }
}

// ---------------------------------------------------------------------------
extern "C" void kernel_launch(void* const* d_in, const int* in_sizes, int n_in,
                              void* d_out, int out_size, void* d_ws, size_t ws_size,
                              hipStream_t stream) {
    const float* emb_user = (const float*)d_in[0];
    const float* emb_item = (const float*)d_in[1];
    const float* beh[4] = {(const float*)d_in[2], (const float*)d_in[3],
                           (const float*)d_in[4], nullptr};  // pv, cart, buy, all
    const int* batch_users = (const int*)d_in[32];

    const size_t ND = (size_t)NTOT * DDIM;

    // ---- workspace layout ----
    char* wbase = (char*)d_ws;
    size_t off = 0;
    auto alloc = [&](size_t bytes) -> void* {
        void* p = wbase + off;
        off = (off + bytes + 255) & ~(size_t)255;
        return p;
    };
    float* bufA = (float*)alloc(ND * sizeof(float));   // ping
    float* bufB = (float*)alloc(ND * sizeof(float));   // pong
    float* acc  = (float*)alloc(ND * sizeof(float));
    float* ub_raw[4];
    float* ib_raw[4];
    for (int b = 0; b < 4; ++b) {
        ub_raw[b] = (float*)alloc((size_t)BBATCH * DDIM * sizeof(float));
        ib_raw[b] = (float*)alloc((size_t)NII * DDIM * sizeof(float));
    }
    _Float16* ubw_h[3];
    _Float16* ib_h[3];
    for (int b = 0; b < 3; ++b) {
        ubw_h[b] = (_Float16*)alloc((size_t)BBATCH * DDIM * sizeof(_Float16));
        ib_h[b]  = (_Float16*)alloc((size_t)NII * DDIM * sizeof(_Float16));
    }

    const int nThreads = 256;
    const int nd4Blocks = (int)((NTOT * (DDIM / 4) + nThreads - 1) / nThreads);
    const int exBlocks = ((BBATCH + NII) * (DDIM / 4) + nThreads - 1) / nThreads;

    // ---- 4 propagations: pv(0), cart(1), buy(2), all(3) ----
    for (int b = 0; b < 4; ++b) {
        const int* rows = (const int*)d_in[5 + 3 * b];
        const int* cols = (const int*)d_in[6 + 3 * b];
        const float* vals = (const float*)d_in[7 + 3 * b];
        int E = in_sizes[5 + 3 * b];

        float* x = bufA;
        float* y = bufB;
        init_kernel<<<nd4Blocks, nThreads, 0, stream>>>(emb_user, emb_item, beh[b], x, acc, y);
        for (int layer = 0; layer < 3; ++layer) {
            long long work = (long long)E * 16;
            int eBlocks = (int)((work + nThreads - 1) / nThreads);
            spmm_kernel<<<eBlocks, nThreads, 0, stream>>>(rows, cols, vals, E, x, y);
            acc_add_zero_kernel<<<nd4Blocks, nThreads, 0, stream>>>(acc, y, x, layer < 2 ? 1 : 0);
            float* t = x; x = y; y = t;
        }
        extract_kernel<<<exBlocks, nThreads, 0, stream>>>(acc, batch_users, ub_raw[b], ib_raw[b]);
    }

    // ---- gate fusion + normalize + weight fold -> f16 GEMM operands ----
    for (int b = 0; b < 3; ++b) {
        const float* guW = (const float*)d_in[17 + 4 * b];
        const float* gub = (const float*)d_in[18 + 4 * b];
        const float* giW = (const float*)d_in[19 + 4 * b];
        const float* gib = (const float*)d_in[20 + 4 * b];
        const float* wv  = (const float*)d_in[29 + b];
        gate_user_kernel<<<BBATCH, DDIM, 0, stream>>>(ub_raw[3], ub_raw[b], guW, gub, wv, ubw_h[b]);
        gate_item_kernel<<<NII, DDIM, 0, stream>>>(ib_raw[3], ib_raw[b], giW, gib, ib_h[b]);
    }

    // ---- likelihood GEMMs via WMMA ----
    float* out = (float*)d_out;
    dim3 ggrid(((NII / 16) + 7) / 8, BBATCH / 16);
    for (int b = 0; b < 3; ++b) {
        wmma_gemm_relu_kernel<<<ggrid, 256, 0, stream>>>(
            ubw_h[b], ib_h[b], out + (size_t)b * BBATCH * NII);
    }
}
